// StructureModule_48928267436136
// MI455X (gfx1250) — matrix operand order repaired and checked by reading, not compiled, not run on hardware
//
#include <hip/hip_runtime.h>

// ---------------- model constants ----------------
#define NRES 512
#define CSd  384
#define CZd  128
#define NHd  12
#define CHd  16
#define PQd  4
#define PVd  8
#define CATD 2112          // H*(C_Z + C_IPA + PV*4)
#define INFM 100000.0f
#define EPSC 1e-8f

#define S1C  0.14433756729740643f   // 1/sqrt(3*C_IPA)
#define S2C  0.57735026918962576f   // sqrt(1/3)
#define HWSC 0.13608276348795434f   // sqrt(1/(3*(PQ*9/2)))

typedef __attribute__((ext_vector_type(16))) __bf16 v16bf;
typedef __attribute__((ext_vector_type(8)))  __bf16 v8bf;
typedef __attribute__((ext_vector_type(8)))  float  v8f;

__device__ __forceinline__ __bf16 f2bf(float x) { return (__bf16)x; }
__device__ __forceinline__ float softplusf(float x) {
  return (x > 20.f) ? x : log1pf(__expf(x));
}

// A fragment (16-bit A 16x32): lane holds K [half*8..+7] and [16+half*8..+7] of its row.
// p must point at rowbase + half*8 (+k0): two contiguous 16B loads.
__device__ __forceinline__ v16bf ld_afrag(const __bf16* p) {
  v8bf x = *(const v8bf*)(p);
  v8bf y = *(const v8bf*)(p + 16);
  return __builtin_shufflevector(x, y, 0, 1, 2, 3, 4, 5, 6, 7, 8, 9, 10, 11, 12, 13, 14, 15);
}
// B fragment (16-bit B 32x16): lane holds K [half*8..+7] and [16+half*8..+7] of its column (W^T row).
__device__ __forceinline__ v16bf ld_bfrag(const __bf16* p) {
  v8bf x = *(const v8bf*)(p);
  v8bf y = *(const v8bf*)(p + 16);
  return __builtin_shufflevector(x, y, 0, 1, 2, 3, 4, 5, 6, 7, 8, 9, 10, 11, 12, 13, 14, 15);
}

// flags: bit1 = relu(out), bit2 = transposed store (out[n*ldo + m]), bit3 = relu on bf16 out only
__device__ __forceinline__ void sm_store_tile(const v8f& acc, int rbase, int nIdx, int M, int N,
                                              const float* bias, const float* res, float* OF,
                                              __bf16* OB, int ldo, int ldres, int flags,
                                              int half) {
  if (nIdx >= N) return;
  float bv = bias ? bias[nIdx] : 0.f;
#pragma unroll
  for (int r = 0; r < 8; ++r) {
    int m = rbase + r + half * 8;   // C/D layout: VGPR r -> M=r / M=8+r
    if (m < M) {
      float v = acc[r] + bv;
      if (res) v += res[(long long)m * ldres + nIdx];
      if (flags & 2) v = fmaxf(v, 0.f);
      long long off = (flags & 4) ? ((long long)nIdx * ldo + m) : ((long long)m * ldo + nIdx);
      if (OF) OF[off] = v;
      if (OB) OB[off] = f2bf((flags & 8) ? fmaxf(v, 0.f) : v);
    }
  }
}

// ------------- bf16 WMMA GEMM, one wave per (MT*16 x NT*16) C tile, batched ---------------
// A: bf16 [M x K] row-major (lda). BT: bf16 [N x K] row-major (ldbt) == B transposed.
// outF (f32) / outB (bf16) optional, row stride ldo. res: f32 residual, stride ldres.
// K must be a multiple of 32.
template <int MT, int NT>
__global__ void sm_gemm_wmma(const __bf16* __restrict__ A, const __bf16* __restrict__ BT,
                             const float* __restrict__ bias, const float* __restrict__ res,
                             float* __restrict__ outF, __bf16* __restrict__ outB,
                             int M, int N, int K, int lda, int ldbt, int ldo, int ldres,
                             int flags, long long aBatch, long long bBatch, long long cBatch) {
  const int lane = threadIdx.x & 31;
  const int half = lane >> 4;
  const int l15  = lane & 15;
  const int row0 = blockIdx.y * (MT * 16);
  const int col0 = blockIdx.x * (NT * 16);
  const __bf16* Ab = A + aBatch * blockIdx.z;
  const __bf16* Bb = BT + bBatch * blockIdx.z;
  float*  OF = outF ? outF + cBatch * blockIdx.z : (float*)0;
  __bf16* OB = outB ? outB + cBatch * blockIdx.z : (__bf16*)0;

  const __bf16* ap[MT];
  const __bf16* bp[NT];
#pragma unroll
  for (int mi = 0; mi < MT; ++mi) {
    int r = row0 + mi * 16 + l15; if (r > M - 1) r = M - 1;   // clamp; stores are guarded
    ap[mi] = Ab + (long long)r * lda + half * 8;
  }
#pragma unroll
  for (int ni = 0; ni < NT; ++ni) {
    int c = col0 + ni * 16 + l15; if (c > N - 1) c = N - 1;
    bp[ni] = Bb + (long long)c * ldbt + half * 8;
  }
  v8f zero = {};
  v8f acc[MT][NT];
#pragma unroll
  for (int mi = 0; mi < MT; ++mi)
#pragma unroll
    for (int ni = 0; ni < NT; ++ni) acc[mi][ni] = zero;

#pragma unroll 2
  for (int k0 = 0; k0 < K; k0 += 32) {
    v16bf fa[MT], fb[NT];
#pragma unroll
    for (int mi = 0; mi < MT; ++mi) fa[mi] = ld_afrag(ap[mi] + k0);
#pragma unroll
    for (int ni = 0; ni < NT; ++ni) fb[ni] = ld_bfrag(bp[ni] + k0);
#pragma unroll
    for (int mi = 0; mi < MT; ++mi)
#pragma unroll
      for (int ni = 0; ni < NT; ++ni)
        acc[mi][ni] = __builtin_amdgcn_wmma_f32_16x16x32_bf16(
            false, fa[mi], false, fb[ni], (short)0, acc[mi][ni], false, false);
  }
#pragma unroll
  for (int mi = 0; mi < MT; ++mi)
#pragma unroll
    for (int ni = 0; ni < NT; ++ni)
      sm_store_tile(acc[mi][ni], row0 + mi * 16, col0 + ni * 16 + l15, M, N, bias, res,
                    OF, OB, ldo, ldres, flags, half);
}

template <int MT, int NT>
static void launch_gemm(hipStream_t stream, const __bf16* A, const __bf16* BT, const float* bias,
                        const float* res, float* outF, __bf16* outB, int M, int N, int K,
                        int lda, int ldbt, int ldo, int ldres, int flags,
                        long long aB, long long bB, long long cB, int batch) {
  dim3 g((N + NT * 16 - 1) / (NT * 16), (M + MT * 16 - 1) / (MT * 16), batch);
  sm_gemm_wmma<MT, NT><<<g, dim3(32), 0, stream>>>(A, BT, bias, res, outF, outB, M, N, K,
                                                   lda, ldbt, ldo, ldres, flags, aB, bB, cB);
}

// ---------------- weight transpose + bf16: WT[n*K+k] = bf16(W[k*N+n]) ----------------
__global__ void sm_wtrans(const float* __restrict__ W, __bf16* __restrict__ WT, int K, int N) {
  long long idx = (long long)blockIdx.x * blockDim.x + threadIdx.x;
  if (idx >= (long long)K * N) return;
  int n = (int)(idx / K), k = (int)(idx % K);
  WT[idx] = f2bf(W[(long long)k * N + n]);
}

// ---- LayerNorm for s rows: f32 out + optional bf16 out + optional relu'd bf16 out ----
__global__ void sm_ln_s(const float* __restrict__ x, const float* __restrict__ g,
                        const float* __restrict__ b, float* __restrict__ outf,
                        __bf16* __restrict__ outb, __bf16* __restrict__ outbR, int C) {
  const long long row = blockIdx.x;
  const float* xr = x + row * C;
  __shared__ float ssum[128], ssq[128];
  float a = 0.f, q = 0.f;
  for (int c = threadIdx.x; c < C; c += 128) { float v = xr[c]; a += v; q += v * v; }
  ssum[threadIdx.x] = a; ssq[threadIdx.x] = q;
  __syncthreads();
  for (int s = 64; s > 0; s >>= 1) {
    if ((int)threadIdx.x < s) { ssum[threadIdx.x] += ssum[threadIdx.x + s];
                                ssq[threadIdx.x]  += ssq[threadIdx.x + s]; }
    __syncthreads();
  }
  float mu  = ssum[0] / C;
  float var = fmaxf(ssq[0] / C - mu * mu, 0.f);
  float inv = rsqrtf(var + 1e-5f);
  for (int c = threadIdx.x; c < C; c += 128) {
    float o = (xr[c] - mu) * inv * g[c] + b[c];
    if (outf)  outf[row * C + c] = o;
    if (outb)  outb[row * C + c] = f2bf(o);
    if (outbR) outbR[row * C + c] = f2bf(fmaxf(o, 0.f));
  }
}

// -------- LayerNorm for z rows (C=128): writes bf16 row-major + bf16 transposed --------
__global__ void sm_ln_z(const float* __restrict__ z, const float* __restrict__ g,
                        const float* __restrict__ b, __bf16* __restrict__ zl,
                        __bf16* __restrict__ zlT) {
  const int row = blockIdx.x;          // i*NRES + j
  const int i = row >> 9, j = row & (NRES - 1);
  const int c = threadIdx.x;           // 128 threads
  const float* xr = z + (long long)row * CZd;
  __shared__ float ssum[128], ssq[128];
  float v = xr[c];
  ssum[c] = v; ssq[c] = v * v;
  __syncthreads();
  for (int s = 64; s > 0; s >>= 1) {
    if (c < s) { ssum[c] += ssum[c + s]; ssq[c] += ssq[c + s]; }
    __syncthreads();
  }
  float mu  = ssum[0] / CZd;
  float var = fmaxf(ssq[0] / CZd - mu * mu, 0.f);
  float inv = rsqrtf(var + 1e-5f);
  float o = (v - mu) * inv * g[c] + b[c];
  zl[(long long)row * CZd + c] = f2bf(o);
  zlT[((long long)i * CZd + c) * NRES + j] = f2bf(o);
}

// ---------------- point transform: out[i,p,:] = R_i @ raw[i,:,p] + t_i ----------------
__global__ void sm_pts_transform(const float* __restrict__ raw, const float* __restrict__ rot,
                                 const float* __restrict__ trans, float* __restrict__ out,
                                 int npts) {
  int idx = blockIdx.x * blockDim.x + threadIdx.x;
  if (idx >= NRES * npts) return;
  int i = idx / npts, p = idx % npts;
  const float* R = rot + i * 9;
  const float* rr = raw + (long long)i * 3 * npts;
  float x = rr[p], y = rr[npts + p], z = rr[2 * npts + p];
  float* o = out + (long long)idx * 3;
  o[0] = R[0] * x + R[1] * y + R[2] * z + trans[i * 3 + 0];
  o[1] = R[3] * x + R[4] * y + R[5] * z + trans[i * 3 + 1];
  o[2] = R[6] * x + R[7] * y + R[8] * z + trans[i * 3 + 2];
}

// ------------- k2T[h][j] = hw_h * sum |k_pts|^2 (q2 cancels in softmax) -------------
__global__ void sm_k2(const float* __restrict__ kvpt, const float* __restrict__ hwp,
                      float* __restrict__ k2T) {
  int idx = blockIdx.x * blockDim.x + threadIdx.x;
  if (idx >= NRES * NHd) return;
  int i = idx / NHd, h = idx % NHd;
  float hw = softplusf(hwp[h]) * HWSC;
  float s = 0.f;
  for (int p = 0; p < PQd; ++p)
    for (int c = 0; c < 3; ++c) {
      float v = kvpt[((long long)i * 144 + h * 12 + p) * 3 + c];
      s += v * v;
    }
  k2T[h * NRES + i] = hw * s;
}

// -------- pack attention operands: u[h][i][32], w[h][j][32] bf16 (16 chan + 12 pt dims) ---
__global__ void sm_build_uw(const float* __restrict__ q, const float* __restrict__ kv,
                            const float* __restrict__ qpt, const float* __restrict__ kvpt,
                            const float* __restrict__ hwp, __bf16* __restrict__ ubuf,
                            __bf16* __restrict__ wbuf) {
  int idx = blockIdx.x * blockDim.x + threadIdx.x;
  if (idx >= NHd * NRES * 32) return;
  int h = idx / (NRES * 32);
  int r = idx % (NRES * 32);
  int i = r / 32, c = r % 32;
  float hw = softplusf(hwp[h]) * HWSC;
  float u = 0.f, w = 0.f;
  if (c < 16) {
    u = q[(long long)i * (NHd * CHd) + h * CHd + c] * S1C;
    w = kv[(long long)i * (NHd * 2 * CHd) + h * 32 + c];
  } else if (c < 28) {
    int tt = c - 16;
    u = qpt[((long long)i * 48 + h * 4 + tt / 3) * 3 + (tt % 3)] * hw;
    w = kvpt[((long long)i * 144 + h * 12 + tt / 3) * 3 + (tt % 3)];
  }
  ubuf[idx] = f2bf(u);
  wbuf[idx] = f2bf(w);
}

// ------ softmax over j: adds j-dependent bias terms, writes bf16 probabilities ------
// bpairT: [h][i*NRES+j] (contiguous in j), k2T: [h][j]
__global__ void sm_softmax_bf(const float* __restrict__ logits, const float* __restrict__ bpairT,
                              const float* __restrict__ k2T, const float* __restrict__ mask,
                              __bf16* __restrict__ out) {
  const int row = blockIdx.x;          // h*NRES + i
  const int h = row >> 9, i = row & (NRES - 1);
  const float* lr = logits + (long long)row * NRES;
  const float* bp = bpairT + (long long)h * NRES * NRES + (long long)i * NRES;
  const float* kp = k2T + (long long)h * NRES;
  __shared__ float red[128];
  const int t = threadIdx.x;
  const float mi = mask[i];
  float v[4];
  float mx = -3.4e38f;
#pragma unroll
  for (int r = 0; r < 4; ++r) {
    int c = t + r * 128;
    float val = lr[c] + S2C * bp[c] - 0.5f * kp[c] + INFM * (mi * mask[c] - 1.f);
    v[r] = val;
    mx = fmaxf(mx, val);
  }
  red[t] = mx; __syncthreads();
  for (int s = 64; s > 0; s >>= 1) { if (t < s) red[t] = fmaxf(red[t], red[t + s]); __syncthreads(); }
  mx = red[0]; __syncthreads();
  float sum = 0.f;
#pragma unroll
  for (int r = 0; r < 4; ++r) { v[r] = __expf(v[r] - mx); sum += v[r]; }
  red[t] = sum; __syncthreads();
  for (int s = 64; s > 0; s >>= 1) { if (t < s) red[t] += red[t + s]; __syncthreads(); }
  float inv = 1.f / red[0];
#pragma unroll
  for (int r = 0; r < 4; ++r) out[(long long)row * NRES + t + r * 128] = f2bf(v[r] * inv);
}

// -------- V||V_pts transposed per head: vcatT[h][c(40)][j] bf16 --------
__global__ void sm_build_vcatT(const float* __restrict__ kv, const float* __restrict__ kvpt,
                               __bf16* __restrict__ vcatT) {
  int idx = blockIdx.x * blockDim.x + threadIdx.x;
  if (idx >= NHd * 40 * NRES) return;
  int h = idx / (40 * NRES);
  int r = idx % (40 * NRES);
  int c = r / NRES, j = r % NRES;
  float v;
  if (c < 16) v = kv[(long long)j * (NHd * 2 * CHd) + h * 32 + 16 + c];
  else { int tt = c - 16;
         v = kvpt[((long long)j * 144 + h * 12 + 4 + tt / 3) * 3 + (tt % 3)]; }
  vcatT[idx] = f2bf(v);
}

// ---- inverse-rotate o_pt, norms, scatter o + o_pt + norm into cat (bf16) ----
__global__ void sm_ipa_post(const float* __restrict__ obuf, const float* __restrict__ rot,
                            const float* __restrict__ trans, __bf16* __restrict__ cat) {
  int i = blockIdx.x;
  int t = threadIdx.x;           // 192 threads
  {
    int h = t >> 4, c = t & 15;
    cat[(long long)i * CATD + h * CHd + c] = f2bf(obuf[((long long)h * NRES + i) * 40 + c]);
  }
  if (t < NHd * PVd) {
    int h = t >> 3, p = t & 7;
    const float* ob = obuf + ((long long)h * NRES + i) * 40 + 16 + p * 3;
    const float* R = rot + i * 9;
    const float* T = trans + i * 3;
    float g0 = ob[0] - T[0], g1 = ob[1] - T[1], g2 = ob[2] - T[2];
    float l0 = R[0] * g0 + R[3] * g1 + R[6] * g2;   // R^T
    float l1 = R[1] * g0 + R[4] * g1 + R[7] * g2;
    float l2 = R[2] * g0 + R[5] * g1 + R[8] * g2;
    float nr = sqrtf(l0 * l0 + l1 * l1 + l2 * l2 + EPSC);
    __bf16* cr = cat + (long long)i * CATD;
    int col = h * PVd + p;
    cr[192 + col] = f2bf(l0);
    cr[288 + col] = f2bf(l1);
    cr[384 + col] = f2bf(l2);
    cr[480 + col] = f2bf(nr);
  }
}

// ---------------- identity frames ----------------
__global__ void sm_init_frames(float* __restrict__ rot, float* __restrict__ trans) {
  int i = blockIdx.x * blockDim.x + threadIdx.x;
  if (i >= NRES) return;
  float* R = rot + i * 9;
  R[0] = 1.f; R[1] = 0.f; R[2] = 0.f;
  R[3] = 0.f; R[4] = 1.f; R[5] = 0.f;
  R[6] = 0.f; R[7] = 0.f; R[8] = 1.f;
  trans[i * 3 + 0] = trans[i * 3 + 1] = trans[i * 3 + 2] = 0.f;
}

// ---------------- backbone frame update from 'upd' (quat + translation) ----------------
__global__ void sm_quat_update(const float* __restrict__ upd, float* __restrict__ rot,
                               float* __restrict__ trans) {
  int i = blockIdx.x * blockDim.x + threadIdx.x;
  if (i >= NRES) return;
  const float* u = upd + i * 6;
  float qa = 1.f, qb = u[0], qc = u[1], qd = u[2];
  float inv = rsqrtf(qa * qa + qb * qb + qc * qc + qd * qd);
  qa *= inv; qb *= inv; qc *= inv; qd *= inv;
  float Rq[9];
  Rq[0] = qa * qa + qb * qb - qc * qc - qd * qd;
  Rq[1] = 2.f * (qb * qc - qa * qd);
  Rq[2] = 2.f * (qb * qd + qa * qc);
  Rq[3] = 2.f * (qb * qc + qa * qd);
  Rq[4] = qa * qa - qb * qb + qc * qc - qd * qd;
  Rq[5] = 2.f * (qc * qd - qa * qb);
  Rq[6] = 2.f * (qb * qd - qa * qc);
  Rq[7] = 2.f * (qc * qd + qa * qb);
  Rq[8] = qa * qa - qb * qb - qc * qc + qd * qd;
  float* R = rot + i * 9;
  float* T = trans + i * 3;
  float Ro[9], Tn[3];
  for (int r = 0; r < 3; ++r) {
    Tn[r] = R[r * 3 + 0] * u[3] + R[r * 3 + 1] * u[4] + R[r * 3 + 2] * u[5] + T[r];
    for (int c = 0; c < 3; ++c)
      Ro[r * 3 + c] = R[r * 3 + 0] * Rq[c] + R[r * 3 + 1] * Rq[3 + c] + R[r * 3 + 2] * Rq[6 + c];
  }
  for (int k = 0; k < 9; ++k) R[k] = Ro[k];
  for (int k = 0; k < 3; ++k) T[k] = Tn[k];
}

// ---------------- write d_out = concat(s, unnorm, ang, frames) ----------------
__global__ void sm_finalize(const float* __restrict__ s, const float* __restrict__ unn,
                            const float* __restrict__ rot, const float* __restrict__ trans,
                            float* __restrict__ out) {
  int i = blockIdx.x, t = threadIdx.x;     // 64 threads
  for (int c = t; c < CSd; c += 64) out[(long long)i * CSd + c] = s[(long long)i * CSd + c];
  if (t < 14) {
    float v = unn[i * 14 + t];
    out[196608 + i * 14 + t] = v;          // unnorm
    int na = t >> 1;
    float x = unn[i * 14 + na * 2], y = unn[i * 14 + na * 2 + 1];
    float inv = rsqrtf(fmaxf(x * x + y * y, EPSC));
    out[203776 + i * 14 + t] = v * inv;    // ang
  }
  if (t < 12)
    out[210944 + i * 12 + t] = (t < 9) ? rot[i * 9 + t] : trans[i * 3 + (t - 9)] * 10.0f;
}

// ============================ host orchestration ============================
extern "C" void kernel_launch(void* const* d_in, const int* in_sizes, int n_in,
                              void* d_out, int out_size, void* d_ws, size_t ws_size,
                              hipStream_t stream) {
  (void)in_sizes; (void)n_in; (void)out_size; (void)ws_size;

  const float* s_in   = (const float*)d_in[0];
  const float* z_in   = (const float*)d_in[1];
  const float* maskp  = (const float*)d_in[2];
  const float* ln_s_g = (const float*)d_in[3];
  const float* ln_s_b = (const float*)d_in[4];
  const float* ln_z_g = (const float*)d_in[5];
  const float* ln_z_b = (const float*)d_in[6];
  const float* lin_w  = (const float*)d_in[7];
  const float* lin_b  = (const float*)d_in[8];
  const float* q_w    = (const float*)d_in[9];
  const float* q_b    = (const float*)d_in[10];
  const float* kv_w   = (const float*)d_in[11];
  const float* kv_b   = (const float*)d_in[12];
  const float* qp_w   = (const float*)d_in[13];
  const float* qp_b   = (const float*)d_in[14];
  const float* kvp_w  = (const float*)d_in[15];
  const float* kvp_b  = (const float*)d_in[16];
  const float* bz_w   = (const float*)d_in[17];
  const float* bz_b   = (const float*)d_in[18];
  const float* hw_p   = (const float*)d_in[19];
  const float* ow_w   = (const float*)d_in[20];
  const float* ow_b   = (const float*)d_in[21];
  const float* lni_g  = (const float*)d_in[22];
  const float* lni_b  = (const float*)d_in[23];
  const float* t1w = (const float*)d_in[24]; const float* t1b = (const float*)d_in[25];
  const float* t2w = (const float*)d_in[26]; const float* t2b = (const float*)d_in[27];
  const float* t3w = (const float*)d_in[28]; const float* t3b = (const float*)d_in[29];
  const float* lnt_g = (const float*)d_in[30]; const float* lnt_b = (const float*)d_in[31];
  const float* bb_w = (const float*)d_in[32]; const float* bb_b = (const float*)d_in[33];
  const float* ai_w = (const float*)d_in[34]; const float* ai_b = (const float*)d_in[35];
  const float* a0_w = (const float*)d_in[36]; const float* a0_b = (const float*)d_in[37];
  const float* b0l1w = (const float*)d_in[38]; const float* b0l1b = (const float*)d_in[39];
  const float* b0l2w = (const float*)d_in[40]; const float* b0l2b = (const float*)d_in[41];
  const float* b1l1w = (const float*)d_in[42]; const float* b1l1b = (const float*)d_in[43];
  const float* b1l2w = (const float*)d_in[44]; const float* b1l2b = (const float*)d_in[45];
  const float* ao_w = (const float*)d_in[46]; const float* ao_b = (const float*)d_in[47];
  float* outp = (float*)d_out;

  // ---- workspace carve-up (256B aligned) ----
  char* wp = (char*)d_ws;
  auto take = [&](size_t bytes) { char* r = wp; wp += (bytes + 255) & ~(size_t)255; return r; };
  __bf16* zl     = (__bf16*)take(2ull * NRES * NRES * CZd);      // z_ln row-major
  __bf16* zlT    = (__bf16*)take(2ull * NRES * CZd * NRES);      // z_ln [i][c][j]
  float*  attn   = (float*)take(4ull * NHd * NRES * NRES);       // logits
  __bf16* attnb  = (__bf16*)take(2ull * NHd * NRES * NRES);      // probabilities
  float*  bpairT = (float*)take(4ull * NHd * NRES * NRES);       // pair bias [h][i*N+j]
  float*  s_cur  = (float*)take(4ull * NRES * CSd);
  __bf16* s_bf   = (__bf16*)take(2ull * NRES * CSd);
  __bf16* s_bfR  = (__bf16*)take(2ull * NRES * CSd);             // relu(s) bf16
  float*  s_ini  = (float*)take(4ull * NRES * CSd);
  __bf16* s_inb  = (__bf16*)take(2ull * NRES * CSd);
  __bf16* s_inbR = (__bf16*)take(2ull * NRES * CSd);             // relu(s_initial) bf16
  float*  qbuf  = (float*)take(4ull * NRES * NHd * CHd);
  float*  kvbuf = (float*)take(4ull * NRES * NHd * 2 * CHd);
  float*  qpraw = (float*)take(4ull * NRES * 144);
  float*  qpt   = (float*)take(4ull * NRES * 144);
  float*  kvpraw= (float*)take(4ull * NRES * 432);
  float*  kvpt  = (float*)take(4ull * NRES * 432);
  float*  k2T   = (float*)take(4ull * NHd * NRES);
  __bf16* ubuf  = (__bf16*)take(2ull * NHd * NRES * 32);
  __bf16* wbuf  = (__bf16*)take(2ull * NHd * NRES * 32);
  __bf16* vcatT = (__bf16*)take(2ull * NHd * 40 * NRES);
  float*  obuf  = (float*)take(4ull * NHd * NRES * 40);
  __bf16* catb  = (__bf16*)take(2ull * NRES * CATD);
  __bf16* tmp1b = (__bf16*)take(2ull * NRES * CSd);
  __bf16* tmp2b = (__bf16*)take(2ull * NRES * CSd);
  float*  upd   = (float*)take(4ull * NRES * 6);
  float*  rot   = (float*)take(4ull * NRES * 9);
  float*  trs   = (float*)take(4ull * NRES * 3);
  float*  abuf  = (float*)take(4ull * NRES * CZd);
  __bf16* abufb = (__bf16*)take(2ull * NRES * CZd);              // relu(a) bf16
  __bf16* atmpb = (__bf16*)take(2ull * NRES * CZd);
  float*  unn   = (float*)take(4ull * NRES * 14);
  // transposed bf16 weights
  __bf16* linT = (__bf16*)take(2ull * CSd * CSd);
  __bf16* qT   = (__bf16*)take(2ull * CSd * 192);
  __bf16* kvT  = (__bf16*)take(2ull * CSd * 384);
  __bf16* qpT  = (__bf16*)take(2ull * CSd * 144);
  __bf16* kvpT = (__bf16*)take(2ull * CSd * 432);
  __bf16* bzT  = (__bf16*)take(2ull * CZd * 12);
  __bf16* owT  = (__bf16*)take(2ull * CATD * 384);
  __bf16* t1T  = (__bf16*)take(2ull * CSd * CSd);
  __bf16* t2T  = (__bf16*)take(2ull * CSd * CSd);
  __bf16* t3T  = (__bf16*)take(2ull * CSd * CSd);
  __bf16* bbT  = (__bf16*)take(2ull * CSd * 6);
  __bf16* aiT  = (__bf16*)take(2ull * CSd * CZd);
  __bf16* a0T  = (__bf16*)take(2ull * CSd * CZd);
  __bf16* b0l1T= (__bf16*)take(2ull * CZd * CZd);
  __bf16* b0l2T= (__bf16*)take(2ull * CZd * CZd);
  __bf16* b1l1T= (__bf16*)take(2ull * CZd * CZd);
  __bf16* b1l2T= (__bf16*)take(2ull * CZd * CZd);
  __bf16* aoT  = (__bf16*)take(2ull * CZd * 14);

  auto wt = [&](const float* W, __bf16* WT, int K, int N) {
    long long tot = (long long)K * N;
    sm_wtrans<<<(unsigned)((tot + 255) / 256), 256, 0, stream>>>(W, WT, K, N);
  };

  // ---- weight prep ----
  wt(lin_w, linT, CSd, CSd);   wt(q_w, qT, CSd, 192);     wt(kv_w, kvT, CSd, 384);
  wt(qp_w, qpT, CSd, 144);     wt(kvp_w, kvpT, CSd, 432); wt(bz_w, bzT, CZd, 12);
  wt(ow_w, owT, CATD, 384);    wt(t1w, t1T, CSd, CSd);    wt(t2w, t2T, CSd, CSd);
  wt(t3w, t3T, CSd, CSd);      wt(bb_w, bbT, CSd, 6);     wt(ai_w, aiT, CSd, CZd);
  wt(a0_w, a0T, CSd, CZd);     wt(b0l1w, b0l1T, CZd, CZd);wt(b0l2w, b0l2T, CZd, CZd);
  wt(b1l1w, b1l1T, CZd, CZd);  wt(b1l2w, b1l2T, CZd, CZd);wt(ao_w, aoT, CZd, 14);

  // ---- setup ----
  sm_ln_s<<<NRES, 128, 0, stream>>>(s_in, ln_s_g, ln_s_b, s_ini, s_inb, s_inbR, CSd);
  sm_ln_z<<<NRES * NRES, 128, 0, stream>>>(z_in, ln_z_g, ln_z_b, zl, zlT);
  launch_gemm<2, 2>(stream, s_inb, linT, lin_b, nullptr, s_cur, s_bf, NRES, CSd, CSd,
                    CSd, CSd, CSd, 0, 0, 0, 0, 0, 1);
  // b_pair = z_ln @ Wb + bb, stored transposed -> bpairT[h][i*N+j]  (skinny N, transposed store)
  launch_gemm<2, 1>(stream, zl, bzT, bz_b, nullptr, bpairT, nullptr, NRES * NRES, NHd, CZd,
                    CZd, CZd, NRES * NRES, 0, 4, 0, 0, 0, 1);
  sm_init_frames<<<2, 256, 0, stream>>>(rot, trs);

  // ---- 8 structure-module iterations ----
  for (int it = 0; it < 8; ++it) {
    launch_gemm<2, 2>(stream, s_bf, qT, q_b, nullptr, qbuf, nullptr, NRES, 192, CSd,
                      CSd, CSd, 192, 0, 0, 0, 0, 0, 1);
    launch_gemm<2, 2>(stream, s_bf, kvT, kv_b, nullptr, kvbuf, nullptr, NRES, 384, CSd,
                      CSd, CSd, 384, 0, 0, 0, 0, 0, 1);
    launch_gemm<2, 2>(stream, s_bf, qpT, qp_b, nullptr, qpraw, nullptr, NRES, 144, CSd,
                      CSd, CSd, 144, 0, 0, 0, 0, 0, 1);
    launch_gemm<2, 2>(stream, s_bf, kvpT, kvp_b, nullptr, kvpraw, nullptr, NRES, 432, CSd,
                      CSd, CSd, 432, 0, 0, 0, 0, 0, 1);
    sm_pts_transform<<<(NRES * 48 + 255) / 256, 256, 0, stream>>>(qpraw, rot, trs, qpt, 48);
    sm_pts_transform<<<(NRES * 144 + 255) / 256, 256, 0, stream>>>(kvpraw, rot, trs, kvpt, 144);
    sm_k2<<<(NRES * NHd + 255) / 256, 256, 0, stream>>>(kvpt, hw_p, k2T);
    sm_build_uw<<<(NHd * NRES * 32 + 255) / 256, 256, 0, stream>>>(qbuf, kvbuf, qpt, kvpt,
                                                                   hw_p, ubuf, wbuf);
    // logits[h] = U[h] @ W[h]^T  (K=32) -> attn f32
    launch_gemm<2, 2>(stream, ubuf, wbuf, nullptr, nullptr, attn, nullptr, NRES, NRES, 32,
                      32, 32, NRES, 0, 0,
                      (long long)NRES * 32, (long long)NRES * 32, (long long)NRES * NRES, NHd);
    sm_softmax_bf<<<NHd * NRES, 128, 0, stream>>>(attn, bpairT, k2T, maskp, attnb);
    sm_build_vcatT<<<(NHd * 40 * NRES + 255) / 256, 256, 0, stream>>>(kvbuf, kvpt, vcatT);
    // o and o_pt: per-head attn[512x512] @ vcat[512x40]
    launch_gemm<2, 2>(stream, attnb, vcatT, nullptr, nullptr, obuf, nullptr, NRES, 40, NRES,
                      NRES, NRES, 40, 0, 0,
                      (long long)NRES * NRES, (long long)40 * NRES, (long long)NRES * 40, NHd);
    // o_pair: per-query-row batch: attn rows (12 x 512) @ z_i (512x128) -> cat[:,576:]
    // (M=12 -> single-row-fragment tiles, no wasted 16-row half)
    launch_gemm<1, 2>(stream, attnb, zlT, nullptr, nullptr, nullptr, catb + 576, NHd, CZd, NRES,
                      NRES * NRES, NRES, CZd, 0, 0,
                      (long long)NRES, (long long)CZd * NRES, (long long)CATD, NRES);
    sm_ipa_post<<<NRES, 192, 0, stream>>>(obuf, rot, trs, catb);
    // s = s + ipa_out(cat)
    launch_gemm<2, 2>(stream, catb, owT, ow_b, s_cur, s_cur, nullptr, NRES, CSd, CATD,
                      CATD, CATD, CSd, CSd, 0, 0, 0, 0, 1);
    sm_ln_s<<<NRES, 128, 0, stream>>>(s_cur, lni_g, lni_b, s_cur, s_bf, nullptr, CSd);
    // transition: s = s + l3(relu(l2(relu(l1(s)))))
    launch_gemm<2, 2>(stream, s_bf, t1T, t1b, nullptr, nullptr, tmp1b, NRES, CSd, CSd,
                      CSd, CSd, CSd, 0, 2, 0, 0, 0, 1);
    launch_gemm<2, 2>(stream, tmp1b, t2T, t2b, nullptr, nullptr, tmp2b, NRES, CSd, CSd,
                      CSd, CSd, CSd, 0, 2, 0, 0, 0, 1);
    launch_gemm<2, 2>(stream, tmp2b, t3T, t3b, s_cur, s_cur, nullptr, NRES, CSd, CSd,
                      CSd, CSd, CSd, CSd, 0, 0, 0, 0, 1);
    // last LN also emits relu(s) bf16 for the angle resnet after the loop
    sm_ln_s<<<NRES, 128, 0, stream>>>(s_cur, lnt_g, lnt_b, s_cur, s_bf, s_bfR, CSd);
    // backbone update (N=6 -> skinny-N tiles)
    launch_gemm<2, 1>(stream, s_bf, bbT, bb_b, nullptr, upd, nullptr, NRES, 6, CSd,
                      CSd, CSd, 6, 0, 0, 0, 0, 0, 1);
    sm_quat_update<<<2, 256, 0, stream>>>(upd, rot, trs);
  }

  // ---- angle resnet (all relu'd operands pre-computed by producers) ----
  launch_gemm<2, 2>(stream, s_bfR, aiT, ai_b, nullptr, abuf, nullptr, NRES, CZd, CSd,
                    CSd, CSd, CZd, 0, 0, 0, 0, 0, 1);
  launch_gemm<2, 2>(stream, s_inbR, a0T, a0_b, abuf, abuf, abufb, NRES, CZd, CSd,
                    CSd, CSd, CZd, CZd, 8, 0, 0, 0, 1);
  launch_gemm<2, 2>(stream, abufb, b0l1T, b0l1b, nullptr, nullptr, atmpb, NRES, CZd, CZd,
                    CZd, CZd, CZd, 0, 2, 0, 0, 0, 1);
  launch_gemm<2, 2>(stream, atmpb, b0l2T, b0l2b, abuf, abuf, abufb, NRES, CZd, CZd,
                    CZd, CZd, CZd, CZd, 8, 0, 0, 0, 1);
  launch_gemm<2, 2>(stream, abufb, b1l1T, b1l1b, nullptr, nullptr, atmpb, NRES, CZd, CZd,
                    CZd, CZd, CZd, 0, 2, 0, 0, 0, 1);
  launch_gemm<2, 2>(stream, atmpb, b1l2T, b1l2b, abuf, abuf, abufb, NRES, CZd, CZd,
                    CZd, CZd, CZd, CZd, 8, 0, 0, 0, 1);
  launch_gemm<2, 1>(stream, abufb, aoT, ao_b, nullptr, unn, nullptr, NRES, 14, CZd,
                    CZd, CZd, 14, 0, 0, 0, 0, 0, 1);

  sm_finalize<<<NRES, 64, 0, stream>>>(s_cur, unn, rot, trs, outp);
}
